// TransformerBlock_37417755083622
// MI455X (gfx1250) — matrix operand order repaired and checked by reading, not compiled, not run on hardware
//
#include <hip/hip_runtime.h>
#include <hip/hip_bf16.h>

// ---------------------------------------------------------------------------
// BERT transformer block forward for MI455X (gfx1250, wave32, WMMA).
// All GEMMs use v_wmma_f32_16x16x32_f16 (f16 inputs, f32 accumulate).
// Attention is flash-style (online softmax) so the S x S score tensor is
// never materialized in HBM.
// ---------------------------------------------------------------------------

#define B_    2
#define S_    2048
#define H_    1024
#define NH_   16
#define DH_   64
#define DFF_  4096
#define MROWS (B_ * S_)   // 4096

typedef __attribute__((ext_vector_type(16))) _Float16 v16h;
typedef __attribute__((ext_vector_type(8)))  float    v8f;

union AFrag { v16h h; unsigned int u[8]; };
union CFrag { v8f  f; float s[8]; };

// A-fragment (16-bit, 16x32) per-VGPR K index (ISA 7.12.2):
//   VGPR v, lane-group g (lane>=16): k = 2*(v&3) + 8*g + 16*(v>>2), pair (k,k+1)
__device__ __forceinline__ int ka_map(int v, int g) {
    return ((v & 3) << 1) + (g << 3) + ((v >> 2) << 4);
}
// B-fragment (16-bit, 32x16): same K order as A-fragment,
//   VGPR v holds pair k = 2*(v&3) + 8*g + 16*(v>>2), (k,k+1); N = lane & 15.
__device__ __forceinline__ int kb_map(int v, int g) {
    return ((v & 3) << 1) + (g << 3) + ((v >> 2) << 4);
}

// ---------------------------------------------------------------------------
// f32 -> f16 elementwise convert (weights)
// ---------------------------------------------------------------------------
__global__ void cvt_f32_f16(const float* __restrict__ in,
                            _Float16* __restrict__ out, int n) {
    int i = blockIdx.x * blockDim.x + threadIdx.x;
    if (i < n) out[i] = (_Float16)in[i];
}

// ---------------------------------------------------------------------------
// LayerNorm (reference: unbiased var ddof=1, divide by (std + eps)), f16 out.
// One 256-thread block per row of 1024.
// ---------------------------------------------------------------------------
__global__ void layernorm_f16(const float* __restrict__ x,
                              const float* __restrict__ gw,
                              const float* __restrict__ bw,
                              _Float16* __restrict__ out) {
    __shared__ float red[256];
    const int row = blockIdx.x;
    const int tid = threadIdx.x;
    const float* xr = x + (size_t)row * H_;

    float v0 = xr[tid], v1 = xr[tid + 256], v2 = xr[tid + 512], v3 = xr[tid + 768];
    red[tid] = v0 + v1 + v2 + v3;
    __syncthreads();
    for (int o = 128; o > 0; o >>= 1) {
        if (tid < o) red[tid] += red[tid + o];
        __syncthreads();
    }
    const float mean = red[0] * (1.0f / (float)H_);
    __syncthreads();

    float d0 = v0 - mean, d1 = v1 - mean, d2 = v2 - mean, d3 = v3 - mean;
    red[tid] = d0 * d0 + d1 * d1 + d2 * d2 + d3 * d3;
    __syncthreads();
    for (int o = 128; o > 0; o >>= 1) {
        if (tid < o) red[tid] += red[tid + o];
        __syncthreads();
    }
    const float var = red[0] * (1.0f / (float)(H_ - 1));
    const float inv = 1.0f / (sqrtf(fmaxf(var, 0.0f)) + 1e-6f);

    _Float16* orow = out + (size_t)row * H_;
    orow[tid]       = (_Float16)(d0 * inv * gw[tid]       + bw[tid]);
    orow[tid + 256] = (_Float16)(d1 * inv * gw[tid + 256] + bw[tid + 256]);
    orow[tid + 512] = (_Float16)(d2 * inv * gw[tid + 512] + bw[tid + 512]);
    orow[tid + 768] = (_Float16)(d3 * inv * gw[tid + 768] + bw[tid + 768]);
}

// ---------------------------------------------------------------------------
// Tiled WMMA GEMM: out[m,n] = sum_k A[m,k] * W[n,k] + bias[n], with epilogues.
// Each wave32 computes a 16(M) x 64(N) strip: 1 A-fragment reused across 4
// B-fragments -> 4 v_wmma per K=32 step. Block = 128 threads = 4 waves.
// ---------------------------------------------------------------------------
enum { MODE_QK = 0, MODE_VT = 1, MODE_GELU = 2, MODE_RES = 3 };

template <int MODE>
__global__ void gemm_wmma(const _Float16* __restrict__ A,
                          const _Float16* __restrict__ W,
                          const float* __restrict__ bias,
                          const float* __restrict__ res,
                          float* __restrict__ outf,
                          _Float16* __restrict__ outh,
                          int M, int N, int K) {
    const int lane = threadIdx.x & 31;
    const int wv   = threadIdx.x >> 5;
    const int g    = lane >> 4;
    const int l15  = lane & 15;

    const int tilesN = N >> 6;                       // 64-wide N strips
    const int id = blockIdx.x * (blockDim.x >> 5) + wv;
    const int tm = id / tilesN;
    const int tn = id % tilesN;

    CFrag c[4];
#pragma unroll
    for (int t = 0; t < 4; ++t)
#pragma unroll
        for (int i = 0; i < 8; ++i) c[t].s[i] = 0.0f;

    const int mrow = tm * 16 + l15;
    const _Float16* arow = A + (size_t)mrow * K;

    for (int k0 = 0; k0 < K; k0 += 32) {
        AFrag a;
#pragma unroll
        for (int v = 0; v < 8; ++v)
            a.u[v] = *reinterpret_cast<const unsigned int*>(arow + k0 + ka_map(v, g));

#pragma unroll
        for (int t = 0; t < 4; ++t) {
            AFrag bm;
            const _Float16* wrow = W + (size_t)(tn * 64 + t * 16 + l15) * K;
#pragma unroll
            for (int v = 0; v < 8; ++v)
                bm.u[v] = *reinterpret_cast<const unsigned int*>(wrow + k0 + kb_map(v, g));
            c[t].f = __builtin_amdgcn_wmma_f32_16x16x32_f16(
                false, a.h, false, bm.h, (short)0, c[t].f, false, false);
        }
    }

    // Epilogue. C layout: VGPR i -> row = i + 8*g, col = lane&15 (+16*t).
#pragma unroll
    for (int t = 0; t < 4; ++t) {
#pragma unroll
        for (int i = 0; i < 8; ++i) {
            const int row = tm * 16 + i + g * 8;
            const int col = tn * 64 + t * 16 + l15;
            float v = c[t].s[i] + bias[col];
            if (MODE == MODE_QK) {
                // out [b, h, s, d] f16
                const int bb = row >> 11, s = row & (S_ - 1);
                const int hh = col >> 6,  d = col & (DH_ - 1);
                outh[(((size_t)(bb * NH_ + hh) * S_ + s) << 6) + d] = (_Float16)v;
            } else if (MODE == MODE_VT) {
                // out [b, h, d, s] f16 (transposed so P*V B-operand is K-contig)
                const int bb = row >> 11, s = row & (S_ - 1);
                const int hh = col >> 6,  d = col & (DH_ - 1);
                outh[((size_t)(bb * NH_ + hh) * DH_ + d) * S_ + s] = (_Float16)v;
            } else if (MODE == MODE_GELU) {
                const float u = v;
                const float gl = 0.5f * u *
                    (1.0f + tanhf(0.7978845608028654f * (u + 0.044715f * u * u * u)));
                outh[(size_t)row * N + col] = (_Float16)gl;
            } else { // MODE_RES: fp32 residual add
                outf[(size_t)row * N + col] = res[(size_t)row * N + col] + v;
            }
        }
    }
}

// ---------------------------------------------------------------------------
// Flash attention: one wave per (b, h, 16-query tile). 32-key blocks.
// Q[b,h,s,d], K[b,h,s,d], Vt[b,h,d,s] all f16; ctx -> [b,s,h*64+d] f16.
// ---------------------------------------------------------------------------
__global__ void attn_flash(const _Float16* __restrict__ Q,
                           const _Float16* __restrict__ Km,
                           const _Float16* __restrict__ Vt,
                           const int* __restrict__ mask,
                           _Float16* __restrict__ ctx) {
    __shared__ __align__(16) _Float16 plds[4][16 * 32];

    const int lane = threadIdx.x & 31;
    const int wv   = threadIdx.x >> 5;
    const int g    = lane >> 4;
    const int l15  = lane & 15;

    const int id = blockIdx.x * 4 + wv;
    const int qt = id & (S_ / 16 - 1);   // 128 query tiles
    const int bh = id >> 7;              // 0..31
    const int bb = bh >> 4;
    const int hh = bh & (NH_ - 1);
    const int q0 = qt * 16;

    const _Float16* Qb = Q  + (size_t)bh * S_ * DH_;
    const _Float16* Kb = Km + (size_t)bh * S_ * DH_;
    const _Float16* Vb = Vt + (size_t)bh * DH_ * S_;
    const int* mb = mask + bb * S_;

    // Q fragments: 16 queries x 64 dh -> two A-fragments (dh 0-31, 32-63)
    AFrag aQ0, aQ1;
#pragma unroll
    for (int v = 0; v < 8; ++v) {
        const int kk = ka_map(v, g);
        const _Float16* qr = Qb + (size_t)(q0 + l15) * DH_;
        aQ0.u[v] = *reinterpret_cast<const unsigned int*>(qr + kk);
        aQ1.u[v] = *reinterpret_cast<const unsigned int*>(qr + 32 + kk);
    }

    CFrag o[4];
    float mrow[8], lrow[8];
#pragma unroll
    for (int i = 0; i < 8; ++i) { mrow[i] = -3.0e38f; lrow[i] = 0.0f; }
#pragma unroll
    for (int t = 0; t < 4; ++t)
#pragma unroll
        for (int i = 0; i < 8; ++i) o[t].s[i] = 0.0f;

    for (int j = 0; j < S_; j += 32) {
        // ---- scores: Q(16x64) x K^T(64x32) -> two 16x16 tiles ----
        CFrag s0, s1;
#pragma unroll
        for (int i = 0; i < 8; ++i) { s0.s[i] = 0.0f; s1.s[i] = 0.0f; }

        AFrag bk;
#pragma unroll
        for (int v = 0; v < 8; ++v)   // keys j..j+15, dh 0-31
            bk.u[v] = *reinterpret_cast<const unsigned int*>(
                Kb + (size_t)(j + l15) * DH_ + kb_map(v, g));
        s0.f = __builtin_amdgcn_wmma_f32_16x16x32_f16(false, aQ0.h, false, bk.h,
                                                      (short)0, s0.f, false, false);
#pragma unroll
        for (int v = 0; v < 8; ++v)   // keys j..j+15, dh 32-63
            bk.u[v] = *reinterpret_cast<const unsigned int*>(
                Kb + (size_t)(j + l15) * DH_ + 32 + kb_map(v, g));
        s0.f = __builtin_amdgcn_wmma_f32_16x16x32_f16(false, aQ1.h, false, bk.h,
                                                      (short)0, s0.f, false, false);
#pragma unroll
        for (int v = 0; v < 8; ++v)   // keys j+16..j+31, dh 0-31
            bk.u[v] = *reinterpret_cast<const unsigned int*>(
                Kb + (size_t)(j + 16 + l15) * DH_ + kb_map(v, g));
        s1.f = __builtin_amdgcn_wmma_f32_16x16x32_f16(false, aQ0.h, false, bk.h,
                                                      (short)0, s1.f, false, false);
#pragma unroll
        for (int v = 0; v < 8; ++v)   // keys j+16..j+31, dh 32-63
            bk.u[v] = *reinterpret_cast<const unsigned int*>(
                Kb + (size_t)(j + 16 + l15) * DH_ + 32 + kb_map(v, g));
        s1.f = __builtin_amdgcn_wmma_f32_16x16x32_f16(false, aQ1.h, false, bk.h,
                                                      (short)0, s1.f, false, false);

        const int mk0 = mb[j + l15];
        const int mk1 = mb[j + 16 + l15];

        // ---- online softmax (row stats per 16-lane group) ----
#pragma unroll
        for (int i = 0; i < 8; ++i) {
            float f0 = s0.s[i] * 0.125f; if (mk0 == 0) f0 = -1.0e9f;
            float f1 = s1.s[i] * 0.125f; if (mk1 == 0) f1 = -1.0e9f;
            float rm = fmaxf(f0, f1);
#pragma unroll
            for (int off = 8; off >= 1; off >>= 1)
                rm = fmaxf(rm, __shfl_xor(rm, off, 32));
            const float mn   = fmaxf(mrow[i], rm);
            const float corr = __expf(mrow[i] - mn);
            const float p0   = __expf(f0 - mn);
            const float p1   = __expf(f1 - mn);
            float rs = p0 + p1;
#pragma unroll
            for (int off = 8; off >= 1; off >>= 1)
                rs += __shfl_xor(rs, off, 32);
            lrow[i] = lrow[i] * corr + rs;
            mrow[i] = mn;
#pragma unroll
            for (int t = 0; t < 4; ++t) o[t].s[i] *= corr;
            const int prow = i + g * 8;
            plds[wv][prow * 32 + l15]      = (_Float16)p0;
            plds[wv][prow * 32 + 16 + l15] = (_Float16)p1;
        }
        __syncthreads();

        // P -> A-fragment (16x32)
        AFrag pA;
#pragma unroll
        for (int v = 0; v < 8; ++v)
            pA.u[v] = *reinterpret_cast<const unsigned int*>(
                &plds[wv][l15 * 32 + ka_map(v, g)]);
        __syncthreads();

        // ---- O += P(16x32) x V(32x64) ----
#pragma unroll
        for (int t = 0; t < 4; ++t) {
            AFrag bv;
#pragma unroll
            for (int v = 0; v < 8; ++v)
                bv.u[v] = *reinterpret_cast<const unsigned int*>(
                    Vb + (size_t)(t * 16 + l15) * S_ + j + kb_map(v, g));
            o[t].f = __builtin_amdgcn_wmma_f32_16x16x32_f16(
                false, pA.h, false, bv.h, (short)0, o[t].f, false, false);
        }
    }

    // ---- normalize and store ctx [b, s, h*64+d] f16 ----
#pragma unroll
    for (int i = 0; i < 8; ++i) {
        const float inv = 1.0f / lrow[i];
        const int row = q0 + i + g * 8;
#pragma unroll
        for (int t = 0; t < 4; ++t) {
            const int col = hh * DH_ + t * 16 + l15;
            ctx[(size_t)(bb * S_ + row) * H_ + col] = (_Float16)(o[t].s[i] * inv);
        }
    }
}

// ---------------------------------------------------------------------------
// Launch: ln1 -> QKV gemms -> flash attn -> out-proj(+res) -> ln2 ->
//         ffn1(gelu) -> ffn2(+res) -> d_out
// ---------------------------------------------------------------------------
extern "C" void kernel_launch(void* const* d_in, const int* in_sizes, int n_in,
                              void* d_out, int out_size, void* d_ws, size_t ws_size,
                              hipStream_t stream) {
    const float* x    = (const float*)d_in[0];
    const int*   mask = (const int*)  d_in[1];
    const float* wq = (const float*)d_in[2];  const float* bq = (const float*)d_in[3];
    const float* wk = (const float*)d_in[4];  const float* bk = (const float*)d_in[5];
    const float* wv = (const float*)d_in[6];  const float* bv = (const float*)d_in[7];
    const float* wo = (const float*)d_in[8];  const float* bo = (const float*)d_in[9];
    const float* w1 = (const float*)d_in[10]; const float* b1 = (const float*)d_in[11];
    const float* w2 = (const float*)d_in[12]; const float* b2 = (const float*)d_in[13];
    const float* g1 = (const float*)d_in[14]; const float* be1 = (const float*)d_in[15];
    const float* g2 = (const float*)d_in[16]; const float* be2 = (const float*)d_in[17];
    float* out = (float*)d_out;

    // workspace layout (byte offsets, all MiB aligned)
    char* ws = (char*)d_ws;
    const size_t MiB = 1024 * 1024;
    _Float16* h16   = (_Float16*)(ws + 0 * MiB);    // 8 MiB  [4096,1024]
    _Float16* wq16  = (_Float16*)(ws + 8 * MiB);    // 2 MiB
    _Float16* wk16  = (_Float16*)(ws + 10 * MiB);   // 2 MiB
    _Float16* wv16  = (_Float16*)(ws + 12 * MiB);   // 2 MiB
    _Float16* wo16  = (_Float16*)(ws + 14 * MiB);   // 2 MiB
    _Float16* w116  = (_Float16*)(ws + 16 * MiB);   // 8 MiB
    _Float16* w216  = (_Float16*)(ws + 24 * MiB);   // 8 MiB
    _Float16* q16   = (_Float16*)(ws + 32 * MiB);   // 8 MiB  [b,h,s,d]
    _Float16* k16   = (_Float16*)(ws + 40 * MiB);   // 8 MiB  [b,h,s,d]
    _Float16* vT16  = (_Float16*)(ws + 48 * MiB);   // 8 MiB  [b,h,d,s]
    _Float16* ctx16 = (_Float16*)(ws + 56 * MiB);   // 8 MiB  [b,s,H]
    float*    out1  = (float*)   (ws + 64 * MiB);   // 16 MiB [4096,1024]
    _Float16* h216  = (_Float16*)(ws + 80 * MiB);   // 8 MiB
    _Float16* ff16  = (_Float16*)(ws + 88 * MiB);   // 32 MiB [4096,4096]
    (void)ws_size; (void)n_in; (void)in_sizes; (void)out_size;

    // 1) weights -> f16
    cvt_f32_f16<<<(H_ * H_) / 256, 256, 0, stream>>>(wq, wq16, H_ * H_);
    cvt_f32_f16<<<(H_ * H_) / 256, 256, 0, stream>>>(wk, wk16, H_ * H_);
    cvt_f32_f16<<<(H_ * H_) / 256, 256, 0, stream>>>(wv, wv16, H_ * H_);
    cvt_f32_f16<<<(H_ * H_) / 256, 256, 0, stream>>>(wo, wo16, H_ * H_);
    cvt_f32_f16<<<(DFF_ * H_) / 256, 256, 0, stream>>>(w1, w116, DFF_ * H_);
    cvt_f32_f16<<<(DFF_ * H_) / 256, 256, 0, stream>>>(w2, w216, DFF_ * H_);

    // 2) LayerNorm1
    layernorm_f16<<<MROWS, 256, 0, stream>>>(x, g1, be1, h16);

    // 3) Q/K/V projections (M=4096, N=1024, K=1024); grid = M/16*N/64/4
    {
        const int grid = (MROWS / 16) * (H_ / 64) / 4;
        gemm_wmma<MODE_QK><<<grid, 128, 0, stream>>>(h16, wq16, bq, nullptr,
                                                     nullptr, q16, MROWS, H_, H_);
        gemm_wmma<MODE_QK><<<grid, 128, 0, stream>>>(h16, wk16, bk, nullptr,
                                                     nullptr, k16, MROWS, H_, H_);
        gemm_wmma<MODE_VT><<<grid, 128, 0, stream>>>(h16, wv16, bv, nullptr,
                                                     nullptr, vT16, MROWS, H_, H_);
    }

    // 4) flash attention: 32 (b,h) * 128 q-tiles = 4096 waves / 4 per block
    attn_flash<<<1024, 128, 0, stream>>>(q16, k16, vT16, mask, ctx16);

    // 5) output projection + residual (fp32): out1 = x + ctx @ wo^T + bo
    {
        const int grid = (MROWS / 16) * (H_ / 64) / 4;
        gemm_wmma<MODE_RES><<<grid, 128, 0, stream>>>(ctx16, wo16, bo, x,
                                                      out1, nullptr, MROWS, H_, H_);
    }

    // 6) LayerNorm2
    layernorm_f16<<<MROWS, 256, 0, stream>>>(out1, g2, be2, h216);

    // 7) FFN up + GELU (M=4096, N=4096, K=1024)
    {
        const int grid = (MROWS / 16) * (DFF_ / 64) / 4;
        gemm_wmma<MODE_GELU><<<grid, 128, 0, stream>>>(h216, w116, b1, nullptr,
                                                       nullptr, ff16, MROWS, DFF_, H_);
    }

    // 8) FFN down + residual -> d_out (M=4096, N=1024, K=4096)
    {
        const int grid = (MROWS / 16) * (H_ / 64) / 4;
        gemm_wmma<MODE_RES><<<grid, 128, 0, stream>>>(ff16, w216, b2, out1,
                                                      out, nullptr, MROWS, H_, DFF_);
    }
}
